// TSRM_Encoder_75307956568513
// MI455X (gfx1250) — matrix operand order repaired and checked by reading, not compiled, not run on hardware
//
#include <hip/hip_runtime.h>
#include <math.h>

typedef __attribute__((ext_vector_type(16))) _Float16 v16h;
typedef __attribute__((ext_vector_type(8)))  _Float16 v8h;
typedef __attribute__((ext_vector_type(8)))  float    v8f;

#define NTOK 384
#define HIDD 1024
#define NEVT 256
#define LCLIP 64
#define FEATD 1024
#define GDIM 64
#define NG 16
#define OUTC 2148
#define LN10K 9.210340371976184f
#define RPB 64          // (n,m)-pair rows per pos-bias block
#define ESTRIDE 1040    // padded f16 row stride (16B aligned, bank-shifted)

static __device__ __forceinline__ v16h hcat(v8h lo, v8h hi) {
  return __builtin_shufflevector(lo, hi, 0,1,2,3,4,5,6,7,8,9,10,11,12,13,14,15);
}

// ---------------- weight f32 -> f16 convert ----------------
__global__ __launch_bounds__(256) void k_cvt(const float* __restrict__ Wp1,
                                             const float* __restrict__ Wp2,
                                             _Float16* __restrict__ W1h,
                                             _Float16* __restrict__ W2h) {
  const int i = blockIdx.x * 256 + threadIdx.x;
  if (i < 1024 * 1024) W1h[i] = (_Float16)Wp1[i];
  if (i < 16 * 1024)   W2h[i] = (_Float16)Wp2[i];
}

// ---------------- event masked mean ----------------
__global__ __launch_bounds__(256) void k_event_mean(const float* __restrict__ feats,
                                                    const int* __restrict__ gidx,
                                                    const float* __restrict__ cmask,
                                                    float* __restrict__ ev) {
  const int e = blockIdx.x, tid = threadIdx.x;
  float a0 = 0, a1 = 0, a2 = 0, a3 = 0, msum = 0;
  for (int l = 0; l < LCLIP; ++l) {
    const float mk = cmask[e * LCLIP + l];
    msum += mk;
    if (mk != 0.0f) {
      const float* row = feats + (size_t)gidx[e * LCLIP + l] * FEATD;
      a0 += mk * row[tid];       a1 += mk * row[tid + 256];
      a2 += mk * row[tid + 512]; a3 += mk * row[tid + 768];
    }
  }
  const float inv = 1.0f / (msum + 1e-5f);
  float* o = ev + (size_t)e * FEATD;
  o[tid] = a0 * inv; o[tid + 256] = a1 * inv; o[tid + 512] = a2 * inv; o[tid + 768] = a3 * inv;
}

// ---------------- event = relu(event @ Wpre^T + b), 4 rows per block ----------------
__global__ __launch_bounds__(256) void k_pre_gemm(const float* __restrict__ A,
                                                  const float* __restrict__ W,
                                                  const float* __restrict__ bias,
                                                  float* __restrict__ outp) {
  const int m0 = blockIdx.x * 4, tid = threadIdx.x;
  __shared__ float As[4][1024];
  for (int i = tid; i < 4 * 1024; i += 256) As[i >> 10][i & 1023] = A[(size_t)m0 * 1024 + i];
  __syncthreads();
  for (int nn = tid; nn < 1024; nn += 256) {
    const float* w = W + (size_t)nn * 1024;
    float c0 = 0, c1 = 0, c2 = 0, c3 = 0;
    for (int k2 = 0; k2 < 1024; ++k2) {
      const float wv = w[k2];
      c0 += As[0][k2] * wv; c1 += As[1][k2] * wv;
      c2 += As[2][k2] * wv; c3 += As[3][k2] * wv;
    }
    const float b = bias[nn];
    outp[(size_t)(m0 + 0) * 1024 + nn] = fmaxf(c0 + b, 0.0f);
    outp[(size_t)(m0 + 1) * 1024 + nn] = fmaxf(c1 + b, 0.0f);
    outp[(size_t)(m0 + 2) * 1024 + nn] = fmaxf(c2 + b, 0.0f);
    outp[(size_t)(m0 + 3) * 1024 + nn] = fmaxf(c3 + b, 0.0f);
  }
}

// ---------------- x = event[expand_idx]; also writes out[:, 0:1024] ----------------
__global__ __launch_bounds__(256) void k_gather_x(const float* __restrict__ ev,
                                                  const int* __restrict__ eidx,
                                                  float* __restrict__ x,
                                                  float* __restrict__ out) {
  const int n = blockIdx.x, tid = threadIdx.x;
  const float* src = ev + (size_t)eidx[n] * HIDD;
  for (int i = tid; i < HIDD; i += 256) {
    const float v2 = src[i];
    x[(size_t)n * HIDD + i] = v2;
    out[(size_t)n * OUTC + i] = v2;
  }
}

// ---------------- per-head q/k/v projections, regrouped into (G=16,N,64) ----------------
__global__ __launch_bounds__(256) void k_qkv(const float* __restrict__ x,
    const float* __restrict__ Wq, const float* __restrict__ bq,
    const float* __restrict__ Wk, const float* __restrict__ bk,
    const float* __restrict__ Wv, const float* __restrict__ bv,
    float* __restrict__ q, float* __restrict__ k, float* __restrict__ v) {
  const int n = blockIdx.x, tid = threadIdx.x;
  __shared__ float xs[1024];
  for (int i = tid; i < 1024; i += 256) xs[i] = x[(size_t)n * 1024 + i];
  __syncthreads();
  for (int ef = tid; ef < 1024; ef += 256) {
    const int h = ef >> 7;
    const float* xr = xs + h * 128;
    const size_t wo = (size_t)ef * 128;
    const float* wq = Wq + wo; const float* wk = Wk + wo; const float* wv = Wv + wo;
    float aq = 0, ak = 0, av = 0;
    for (int d0 = 0; d0 < 128; ++d0) {
      const float xv = xr[d0];
      aq += xv * wq[d0]; ak += xv * wk[d0]; av += xv * wv[d0];
    }
    const size_t o = ((size_t)(ef >> 6) * NTOK + n) * GDIM + (ef & 63);
    q[o] = aq + bq[ef]; k[o] = ak + bk[ef]; v[o] = av + bv[ef];
  }
}

// ---------------- scores[g][n][m] = q.k / 8 ----------------
__global__ __launch_bounds__(128) void k_scores(const float* __restrict__ q,
                                                const float* __restrict__ k,
                                                float* __restrict__ scores) {
  const int n = blockIdx.x, g = blockIdx.y, tid = threadIdx.x;
  __shared__ float qs[GDIM];
  if (tid < GDIM) qs[tid] = q[((size_t)g * NTOK + n) * GDIM + tid];
  __syncthreads();
  for (int m = tid; m < NTOK; m += 128) {
    const float* kr = k + ((size_t)g * NTOK + m) * GDIM;
    float acc = 0.0f;
    for (int d = 0; d < GDIM; ++d) acc += qs[d] * kr[d];
    scores[((size_t)g * NTOK + n) * NTOK + m] = acc * 0.125f;
  }
}

// ---- fused sinusoid -> MLP(1024->1024->16) -> log, += scores (WMMA f16, 64 rows/block) ----
__global__ __launch_bounds__(512) void k_posbias(const float* __restrict__ bbox,
                                                 const _Float16* __restrict__ W1h,
                                                 const float* __restrict__ bp1,
                                                 const _Float16* __restrict__ W2h,
                                                 const float* __restrict__ bp2,
                                                 float* __restrict__ scores) {
  __shared__ _Float16 embS[RPB][ESTRIDE];    // 133,120 B
  __shared__ _Float16 pos1S[RPB][ESTRIDE];   // 133,120 B
  __shared__ float dcS[RPB], dlS[RPB];
  __shared__ float freqS[256];

  const int tid  = threadIdx.x;
  const int wave = tid >> 5;
  const int lane = tid & 31;
  const int mrow = lane & 15;
  const int hi   = lane >> 4;
  const int klo  = hi * 8;

  const int r0 = blockIdx.x * RPB;           // 64 consecutive (n,m) pairs; same n (384%64==0)
  const int n  = r0 / NTOK;
  const int m0 = r0 % NTOK;
  const float CF = LN10K / 256.0f;

  // per-block precompute: dc/dl per row, frequency factors
  if (tid < RPB) {
    const float s_n = bbox[2 * n], e_n = bbox[2 * n + 1];
    const float cen_n = 0.5f * (s_n + e_n);
    const float len_n = fmaxf(e_n - s_n, 0.1f);
    const int m = m0 + tid;
    const float s_m = bbox[2 * m], e_m = bbox[2 * m + 1];
    const float cen_m = 0.5f * (s_m + e_m);
    const float len_m = fmaxf(e_m - s_m, 0.1f);
    dcS[tid] = (cen_n - cen_m) / len_n;
    dlS[tid] = __logf(len_m / len_n);
  }
  if (tid < 256) freqS[tid] = 100.0f * __expf(-CF * (float)tid);
  __syncthreads();

  // phase 0: sinusoid embedding for 64 rows into LDS (f16)
  for (int t = tid; t < RPB * 512; t += 512) {
    const int row = t >> 9;
    const int p = t & 511;
    const int c = p >> 8;
    const int f = p & 255;
    const float val = (c == 0) ? dcS[row] : dlS[row];
    const float dv = val * freqS[f];
    float sv, cv;
    __sincosf(dv, &sv, &cv);
    embS[row][c * 512 + f]       = (_Float16)sv;
    embS[row][c * 512 + 256 + f] = (_Float16)cv;
  }
  __syncthreads();

  // phase 1: layer 1 (1024->1024). 16 waves x 4 passes of 16 cols; 4 M-tiles share each B frag.
  for (int p = 0; p < 4; ++p) {
    const int ncol = p * 256 + wave * 16;
    const int e = ncol + mrow;                 // this lane's output column (Wp1 row)
    const _Float16* __restrict__ wrow = W1h + (size_t)e * 1024;
    v8f ac0 = {}, ac1 = {}, ac2 = {}, ac3 = {};
    for (int k0 = 0; k0 < 1024; k0 += 32) {
      const v16h b = hcat(*(const v8h*)&wrow[k0 + hi * 8],
                          *(const v8h*)&wrow[k0 + 16 + hi * 8]);
      v16h a;
      a = hcat(*(const v8h*)&embS[mrow][k0 + klo],      *(const v8h*)&embS[mrow][k0 + 16 + klo]);
      ac0 = __builtin_amdgcn_wmma_f32_16x16x32_f16(false, a, false, b, (short)0, ac0, false, false);
      a = hcat(*(const v8h*)&embS[16 + mrow][k0 + klo], *(const v8h*)&embS[16 + mrow][k0 + 16 + klo]);
      ac1 = __builtin_amdgcn_wmma_f32_16x16x32_f16(false, a, false, b, (short)0, ac1, false, false);
      a = hcat(*(const v8h*)&embS[32 + mrow][k0 + klo], *(const v8h*)&embS[32 + mrow][k0 + 16 + klo]);
      ac2 = __builtin_amdgcn_wmma_f32_16x16x32_f16(false, a, false, b, (short)0, ac2, false, false);
      a = hcat(*(const v8h*)&embS[48 + mrow][k0 + klo], *(const v8h*)&embS[48 + mrow][k0 + 16 + klo]);
      ac3 = __builtin_amdgcn_wmma_f32_16x16x32_f16(false, a, false, b, (short)0, ac3, false, false);
    }
    const float bias = bp1[e];
    const int col = ncol + mrow;
    #pragma unroll
    for (int j = 0; j < 8; ++j) {
      const int rr = j + hi * 8;
      pos1S[rr][col]      = (_Float16)fmaxf(ac0[j] + bias, 0.0f);
      pos1S[16 + rr][col] = (_Float16)fmaxf(ac1[j] + bias, 0.0f);
      pos1S[32 + rr][col] = (_Float16)fmaxf(ac2[j] + bias, 0.0f);
      pos1S[48 + rr][col] = (_Float16)fmaxf(ac3[j] + bias, 0.0f);
    }
  }
  __syncthreads();

  // phase 2: layer 2 (1024->16). Waves 0..3, one M-tile each (tiny tail).
  if (wave < 4) {
    const int rbase = wave * 16;
    const _Float16* __restrict__ w2 = W2h + (size_t)mrow * 1024;
    v8f acc = {};
    for (int k0 = 0; k0 < 1024; k0 += 32) {
      const v16h a = hcat(*(const v8h*)&pos1S[rbase + mrow][k0 + klo],
                          *(const v8h*)&pos1S[rbase + mrow][k0 + 16 + klo]);
      const v16h b = hcat(*(const v8h*)&w2[k0 + hi * 8],
                          *(const v8h*)&w2[k0 + 16 + hi * 8]);
      acc = __builtin_amdgcn_wmma_f32_16x16x32_f16(false, a, false, b, (short)0, acc, false, false);
    }
    const int g = mrow;                        // D col = group index
    const float bias = bp2[g];
    #pragma unroll
    for (int j = 0; j < 8; ++j) {
      const int row = rbase + j + hi * 8;      // pair row within block
      const float pv = fmaxf(acc[j] + bias, 0.0f);
      const float lb = logf(fmaxf(pv, 1e-6f));
      scores[((size_t)g * NTOK + n) * NTOK + (m0 + row)] += lb;
    }
  }
}

// ---------------- masked softmax + att @ v -> out[:, 1024:2048] ----------------
__global__ __launch_bounds__(128) void k_attn_out(const float* __restrict__ scores,
                                                  const float* __restrict__ bmask,
                                                  const float* __restrict__ v,
                                                  float* __restrict__ out) {
  const int n = blockIdx.x, g = blockIdx.y, tid = threadIdx.x;
  __shared__ float att[NTOK];
  __shared__ float red[128];
  const size_t rowoff = ((size_t)g * NTOK + n) * NTOK;
  float lm = -1e30f;
  for (int m = tid; m < NTOK; m += 128) {
    float s = scores[rowoff + m];
    if (bmask[(size_t)n * NTOK + m] <= 0.0f) s = -1.0e9f;
    att[m] = s;
    lm = fmaxf(lm, s);
  }
  red[tid] = lm; __syncthreads();
  for (int st = 64; st > 0; st >>= 1) { if (tid < st) red[tid] = fmaxf(red[tid], red[tid + st]); __syncthreads(); }
  const float mx = red[0];
  __syncthreads();
  float ls = 0.0f;
  for (int m = tid; m < NTOK; m += 128) { const float p = __expf(att[m] - mx); att[m] = p; ls += p; }
  red[tid] = ls; __syncthreads();
  for (int st = 64; st > 0; st >>= 1) { if (tid < st) red[tid] += red[tid + st]; __syncthreads(); }
  const float inv = 1.0f / red[0];
  __syncthreads();
  for (int m = tid; m < NTOK; m += 128) att[m] *= inv;
  __syncthreads();
  const int d = tid & 63, half = tid >> 6;
  float acc = 0.0f;
  const int mbeg = half * 192;
  for (int m = mbeg; m < mbeg + 192; ++m) acc += att[m] * v[((size_t)g * NTOK + m) * GDIM + d];
  red[tid] = acc; __syncthreads();
  if (tid < 64) out[(size_t)n * OUTC + HIDD + g * GDIM + d] = red[tid] + red[tid + 64];
}

// ---------------- e100 sinusoid of bbox -> out[:, 2048:2148] ----------------
__global__ __launch_bounds__(128) void k_e100(const float* __restrict__ bbox,
                                              float* __restrict__ out) {
  const int n = blockIdx.x, t = threadIdx.x;
  if (t >= 100) return;
  const int c = t / 50, rem = t % 50;
  const int isc = rem / 25, f = rem % 25;
  const float dv = 100.0f * bbox[2 * n + c] * expf(-(LN10K * 0.04f) * (float)f);
  out[(size_t)n * OUTC + 2048 + t] = isc ? cosf(dv) : sinf(dv);
}

extern "C" void kernel_launch(void* const* d_in, const int* in_sizes, int n_in,
                              void* d_out, int out_size, void* d_ws, size_t ws_size,
                              hipStream_t stream) {
  (void)in_sizes; (void)n_in; (void)out_size; (void)ws_size;
  const float* feats = (const float*)d_in[0];
  const int*   gidx  = (const int*)d_in[1];
  const float* cmask = (const float*)d_in[2];
  const int*   eidx  = (const int*)d_in[3];
  const float* bmask = (const float*)d_in[4];
  const float* bbox  = (const float*)d_in[5];
  const float* Wpre  = (const float*)d_in[6];
  const float* bpre  = (const float*)d_in[7];
  const float* Wq    = (const float*)d_in[8];
  const float* bq    = (const float*)d_in[9];
  const float* Wk    = (const float*)d_in[10];
  const float* bk    = (const float*)d_in[11];
  const float* Wv    = (const float*)d_in[12];
  const float* bv    = (const float*)d_in[13];
  const float* Wp1   = (const float*)d_in[14];
  const float* bp1   = (const float*)d_in[15];
  const float* Wp2   = (const float*)d_in[16];
  const float* bp2   = (const float*)d_in[17];
  float* out = (float*)d_out;

  char* ws = (char*)d_ws;
  float*    event_pre = (float*)(ws + 0);            // 256x1024 f32  (1,048,576 B)
  float*    event_act = (float*)(ws + 1048576);      // 256x1024 f32
  float*    x         = (float*)(ws + 2097152);      // 384x1024 f32  (1,572,864 B)
  float*    qb        = (float*)(ws + 3670016);      // 16x384x64 f32
  float*    kb        = (float*)(ws + 5242880);      // 16x384x64 f32
  float*    vb        = (float*)(ws + 6815744);      // 16x384x64 f32
  float*    scores    = (float*)(ws + 8388608);      // 16x384x384 f32 (9,437,184 B)
  _Float16* W1h       = (_Float16*)(ws + 17825792);  // 1024x1024 f16 (2,097,152 B)
  _Float16* W2h       = (_Float16*)(ws + 19922944);  // 16x1024 f16   (32,768 B)

  k_cvt<<<4096, 256, 0, stream>>>(Wp1, Wp2, W1h, W2h);
  k_event_mean<<<NEVT, 256, 0, stream>>>(feats, gidx, cmask, event_pre);
  k_pre_gemm<<<NEVT / 4, 256, 0, stream>>>(event_pre, Wpre, bpre, event_act);
  k_gather_x<<<NTOK, 256, 0, stream>>>(event_act, eidx, x, out);
  k_qkv<<<NTOK, 256, 0, stream>>>(x, Wq, bq, Wk, bk, Wv, bv, qb, kb, vb);
  k_scores<<<dim3(NTOK, NG), 128, 0, stream>>>(qb, kb, scores);
  k_posbias<<<(NTOK * NTOK) / RPB, 512, 0, stream>>>(bbox, W1h, bp1, W2h, bp2, scores);
  k_attn_out<<<dim3(NTOK, NG), 128, 0, stream>>>(scores, bmask, vb, out);
  k_e100<<<NTOK, 128, 0, stream>>>(bbox, out);
}
